// ContextFuseBlock_42691974922624
// MI455X (gfx1250) — matrix operand order repaired and checked by reading, not compiled, not run on hardware
//
#include <hip/hip_runtime.h>

#define DEV __device__ __forceinline__

typedef __attribute__((ext_vector_type(16))) __bf16 bf16x16;
typedef __attribute__((ext_vector_type(8)))  float   f32x8;

union FragB16 { bf16x16 v; uint4 q[2]; };

DEV f32x8 vzero8() { f32x8 z;
#pragma unroll
  for (int i = 0; i < 8; ++i) z[i] = 0.f;
  return z; }

DEV f32x8 wmma_bf16(bf16x16 a, bf16x16 b, f32x8 c) {
  return __builtin_amdgcn_wmma_f32_16x16x32_bf16(false, a, false, b, (short)0, c, false, false);
}

DEV float gelu_exact(float x) { return 0.5f * x * (1.f + erff(x * 0.7071067811865475f)); }

// CDNA5 async global->LDS copy (tracked by ASYNCcnt, bypasses VGPRs).
// LDS destination address = low 32 bits of the generic pointer (aperture rule).
DEV void async_load_b128(void* lds_dst, const void* gsrc) {
  asm volatile("global_load_async_to_lds_b128 %0, %1, off"
               :: "v"((unsigned)(unsigned long long)lds_dst), "v"(gsrc)
               : "memory");
}
DEV void wait_async0() { asm volatile("s_wait_asynccnt 0x0" ::: "memory"); }
DEV void wait_async3() { asm volatile("s_wait_asynccnt 0x3" ::: "memory"); }

// ---------------------------------------------------------------------------
// Weight prep: cast f32 (K x N, row-major) -> bf16 transposed (N x K)
// ---------------------------------------------------------------------------
__global__ void transpose_cast_kernel(const float* __restrict__ src, __bf16* __restrict__ dst,
                                      int K, int N) {
  int idx = blockIdx.x * 256 + threadIdx.x;
  if (idx >= K * N) return;
  int k = idx / N, n = idx % N;
  dst[(size_t)n * K + k] = (__bf16)src[idx];
}

// ---------------------------------------------------------------------------
// LayerNorm + cast to bf16. One block per token row, 256 threads, C = 256.
// ---------------------------------------------------------------------------
__global__ void ln_cast_kernel(const float* __restrict__ x, const float* __restrict__ g,
                               const float* __restrict__ bta, __bf16* __restrict__ out) {
  const int row = blockIdx.x, tid = threadIdx.x;
  const int wave = tid >> 5, lane = tid & 31;
  float v = x[(size_t)row * 256 + tid];
  float s = v, sq = v * v;
#pragma unroll
  for (int m = 1; m < 32; m <<= 1) { s += __shfl_xor(s, m); sq += __shfl_xor(sq, m); }
  __shared__ float rs[8], rq[8];
  if (lane == 0) { rs[wave] = s; rq[wave] = sq; }
  __syncthreads();
  if (wave == 0) {
    float a = (lane < 8) ? rs[lane] : 0.f;
    float b = (lane < 8) ? rq[lane] : 0.f;
#pragma unroll
    for (int m = 1; m < 8; m <<= 1) { a += __shfl_xor(a, m); b += __shfl_xor(b, m); }
    if (lane == 0) { rs[0] = a; rq[0] = b; }
  }
  __syncthreads();
  float mean = rs[0] * (1.f / 256.f);
  float var  = rq[0] * (1.f / 256.f) - mean * mean;
  float rstd = rsqrtf(var + 1e-5f);
  out[(size_t)row * 256 + tid] = (__bf16)((v - mean) * rstd * g[tid] + bta[tid]);
}

// ---------------------------------------------------------------------------
// WMMA GEMM: out = A(bf16, MxK) @ Bt(bf16, stored NxK) + bias, epilogue EPI.
// Block tile 128x64, BK=32, 8 waves x (32x32). Double-buffered LDS staged
// with global_load_async_to_lds_b128; pipeline synchronized on ASYNCcnt.
// EPI: 1 store bf16 | 2 gelu->bf16 | 3 f32 + resA | 4 s2*(acc+b)+s1*resA+resB
// ---------------------------------------------------------------------------
template <int EPI>
__global__ void gemm128x64(const __bf16* __restrict__ A, int lda,
                           const __bf16* __restrict__ Bt, int K,
                           const float* __restrict__ bias, int N,
                           void* __restrict__ outp,
                           const float* __restrict__ resA,
                           const float* __restrict__ resB,
                           const float* __restrict__ s1p,
                           const float* __restrict__ s2p) {
  __shared__ __align__(16) __bf16 sA[2][128 * 40];
  __shared__ __align__(16) __bf16 sB[2][64 * 40];
  const int tid = threadIdx.x, lane = tid & 31, wave = tid >> 5;
  const int wm = wave >> 1, wn = wave & 1;
  const int l15 = lane & 15, hi = lane >> 4;
  const int bm0 = blockIdx.y * 128, bn0 = blockIdx.x * 64;

  // Per-thread staging geometry (3 async b128 per wave per stage).
  const int uA0 = tid * 2, uA1 = uA0 + 1;
  const int rowA0 = uA0 >> 2, c4A0 = uA0 & 3;
  const int rowA1 = uA1 >> 2, c4A1 = uA1 & 3;
  const int rowB = tid >> 2, c4B = tid & 3;
  const __bf16* gA0 = A + (size_t)(bm0 + rowA0) * lda + c4A0 * 8;
  const __bf16* gA1 = A + (size_t)(bm0 + rowA1) * lda + c4A1 * 8;
  const __bf16* gB  = Bt + (size_t)(bn0 + rowB) * K + c4B * 8;
  const int oA0 = rowA0 * 40 + c4A0 * 8;
  const int oA1 = rowA1 * 40 + c4A1 * 8;
  const int oB  = rowB * 40 + c4B * 8;

  auto stage = [&](int ib, int kt) {
    async_load_b128(&sA[ib][oA0], gA0 + kt);
    async_load_b128(&sA[ib][oA1], gA1 + kt);
    async_load_b128(&sB[ib][oB],  gB + kt);
  };

  f32x8 acc[2][2];
#pragma unroll
  for (int i = 0; i < 2; i++)
#pragma unroll
    for (int j = 0; j < 2; j++) acc[i][j] = vzero8();

  stage(0, 0);
  int ib = 0;
  for (int kt = 0; kt < K; kt += 32, ib ^= 1) {
    const bool more = (kt + 32) < K;
    if (more) {
      stage(ib ^ 1, kt + 32);  // prefetch next tile into other buffer
      wait_async3();           // first 3 (current stage) have landed in LDS
    } else {
      wait_async0();           // drain
    }
    __syncthreads();
    const __bf16* pA = sA[ib];
    const __bf16* pB = sB[ib];
    FragB16 af[2], bf[2];
#pragma unroll
    for (int t = 0; t < 2; t++) {
      int m = wm * 32 + t * 16 + l15;
      af[t].q[0] = *(const uint4*)(pA + m * 40 + hi * 8);
      af[t].q[1] = *(const uint4*)(pA + m * 40 + hi * 8 + 16);
      int n = wn * 32 + t * 16 + l15;
      bf[t].q[0] = *(const uint4*)(pB + n * 40 + hi * 8);
      bf[t].q[1] = *(const uint4*)(pB + n * 40 + hi * 8 + 16);
    }
#pragma unroll
    for (int tm = 0; tm < 2; tm++)
#pragma unroll
      for (int tn = 0; tn < 2; tn++)
        acc[tm][tn] = wmma_bf16(af[tm].v, bf[tn].v, acc[tm][tn]);
    __syncthreads();  // all waves done reading before buffer is re-staged
  }

  float s1 = 0.f, s2 = 0.f;
  if (EPI == 4) { s1 = s1p[0]; s2 = s2p[0]; }
#pragma unroll
  for (int tm = 0; tm < 2; tm++)
#pragma unroll
    for (int tn = 0; tn < 2; tn++)
#pragma unroll
      for (int r = 0; r < 8; r++) {
        int m = bm0 + wm * 32 + tm * 16 + r + hi * 8;
        int n = bn0 + wn * 32 + tn * 16 + l15;
        size_t idx = (size_t)m * N + n;
        float v = acc[tm][tn][r] + bias[n];
        if (EPI == 1)       ((__bf16*)outp)[idx] = (__bf16)v;
        else if (EPI == 2)  ((__bf16*)outp)[idx] = (__bf16)gelu_exact(v);
        else if (EPI == 3)  ((float*)outp)[idx] = v + resA[idx];
        else                ((float*)outp)[idx] = s2 * v + s1 * resA[idx] + resB[idx];
      }
}

// ---------------------------------------------------------------------------
// Fused (CSWin-windowed / global) attention, flash-style online softmax.
// L*DH == 8192 for both configs: <DH=32, L=256> global, <DH=64, L=128> LePE.
// One block per (window, head); L/16 waves; each wave owns 16 query rows.
// K tile staged via async-to-LDS; V transposed through VGPRs (scatter).
// ---------------------------------------------------------------------------
template <int DH, int L, bool OUT_BF16>
__global__ void attn_kernel(const __bf16* __restrict__ Qg, const __bf16* __restrict__ Kg,
                            const __bf16* __restrict__ Vg, void* __restrict__ outp,
                            int nheads, int hsp, int wsp, float scale) {
  constexpr int C = 256, RESI = 16, NTOK = 256;
  constexpr int QSTR = DH + 8;     // padded K-row stride (elements)
  constexpr int VSTR = L + 8;      // padded Vt-row stride
  constexpr int NW = L / 16;       // waves per block
  constexpr int KF = DH / 32;      // A fragments along dh
  constexpr int DT = DH / 16;      // output tiles along dh
  __shared__ __align__(16) __bf16 sK[L * QSTR];
  __shared__ __align__(16) __bf16 sVt[DH * VSTR];
  __shared__ __align__(16) __bf16 sP[NW * 16 * 40];

  const int nw_w = RESI / wsp;
  const int nwin = (RESI / hsp) * nw_w;
  const int bid = blockIdx.x;
  const int head = bid % nheads;
  const int win = bid / nheads;
  const int b = win / nwin, wloc = win % nwin;
  const int h0 = (wloc / nw_w) * hsp, w0 = (wloc % nw_w) * wsp;
  const int tid = threadIdx.x;

  // Stage K rows (token-major, async DMA) and V transposed (dh-major).
  for (int idx = tid; idx < L * (DH / 8); idx += 2 * L) {
    int p = idx / (DH / 8);
    int d8 = (idx % (DH / 8)) * 8;
    int n = (h0 + p / wsp) * RESI + (w0 + p % wsp);
    size_t g = ((size_t)b * NTOK + n) * C + head * DH + d8;
    async_load_b128(sK + p * QSTR + d8, Kg + g);
    uint4 vv = *(const uint4*)(Vg + g);
    const __bf16* ve = (const __bf16*)&vv;
#pragma unroll
    for (int t = 0; t < 8; t++) sVt[(d8 + t) * VSTR + p] = ve[t];
  }
  wait_async0();
  __syncthreads();

  const int wave = tid >> 5, lane = tid & 31;
  const int l15 = lane & 15, hi = lane >> 4;
  const int rowbase = wave * 16;

  // Q fragments straight from global in native A-layout (registers all loop).
  FragB16 qf[KF];
  {
    int p = rowbase + l15;
    int n = (h0 + p / wsp) * RESI + (w0 + p % wsp);
    const __bf16* qrow = Qg + ((size_t)b * NTOK + n) * C + head * DH;
    int k0 = hi * 8;
#pragma unroll
    for (int f = 0; f < KF; f++) {
      qf[f].q[0] = *(const uint4*)(qrow + f * 32 + k0);
      qf[f].q[1] = *(const uint4*)(qrow + f * 32 + k0 + 16);
    }
  }

  float mrun[8], lrun[8];
#pragma unroll
  for (int r = 0; r < 8; r++) { mrun[r] = -3.0e38f; lrun[r] = 0.f; }
  f32x8 oacc[DT];
#pragma unroll
  for (int t = 0; t < DT; t++) oacc[t] = vzero8();

  __bf16* pw = sP + wave * 640;  // 16 x 40 per-wave P tile

  for (int kc = 0; kc < L; kc += 32) {
    // S tile: 16 rows x 32 keys
    f32x8 s0 = vzero8(), s1 = vzero8();
#pragma unroll
    for (int f = 0; f < KF; f++) {
      FragB16 bf0, bf1;
      int kb = hi * 8 + f * 32;
      int n0 = (kc + l15) * QSTR, n1 = (kc + 16 + l15) * QSTR;
      bf0.q[0] = *(const uint4*)(sK + n0 + kb);
      bf0.q[1] = *(const uint4*)(sK + n0 + kb + 16);
      bf1.q[0] = *(const uint4*)(sK + n1 + kb);
      bf1.q[1] = *(const uint4*)(sK + n1 + kb + 16);
      s0 = wmma_bf16(qf[f].v, bf0.v, s0);
      s1 = wmma_bf16(qf[f].v, bf1.v, s1);
    }
    // Online softmax (row reductions across the 16-lane half-wave)
    float p0[8], p1[8];
#pragma unroll
    for (int r = 0; r < 8; r++) {
      float a = s0[r] * scale, bb = s1[r] * scale;
      float mx = fmaxf(a, bb);
      mx = fmaxf(mx, __shfl_xor(mx, 1));
      mx = fmaxf(mx, __shfl_xor(mx, 2));
      mx = fmaxf(mx, __shfl_xor(mx, 4));
      mx = fmaxf(mx, __shfl_xor(mx, 8));
      float mnew = fmaxf(mrun[r], mx);
      float alpha = __expf(mrun[r] - mnew);
      float e0 = __expf(a - mnew), e1 = __expf(bb - mnew);
      float rsum = e0 + e1;
      rsum += __shfl_xor(rsum, 1);
      rsum += __shfl_xor(rsum, 2);
      rsum += __shfl_xor(rsum, 4);
      rsum += __shfl_xor(rsum, 8);
      lrun[r] = lrun[r] * alpha + rsum;
      mrun[r] = mnew;
      p0[r] = e0; p1[r] = e1;
#pragma unroll
      for (int t = 0; t < DT; t++) oacc[t][r] *= alpha;
    }
    // C-layout -> A-layout transpose of P via per-wave LDS bounce
#pragma unroll
    for (int r = 0; r < 8; r++) {
      int m = r + hi * 8;
      pw[m * 40 + l15] = (__bf16)p0[r];
      pw[m * 40 + 16 + l15] = (__bf16)p1[r];
    }
    asm volatile("s_wait_dscnt 0x0" ::: "memory");
    FragB16 pf;
    {
      int k0 = hi * 8;
      pf.q[0] = *(const uint4*)(pw + l15 * 40 + k0);
      pf.q[1] = *(const uint4*)(pw + l15 * 40 + k0 + 16);
    }
    // O += P @ V
#pragma unroll
    for (int t = 0; t < DT; t++) {
      FragB16 vf;
      int dn = (t * 16 + l15) * VSTR;
      int kb2 = kc + hi * 8;
      vf.q[0] = *(const uint4*)(sVt + dn + kb2);
      vf.q[1] = *(const uint4*)(sVt + dn + kb2 + 16);
      oacc[t] = wmma_bf16(pf.v, vf.v, oacc[t]);
    }
  }

  float inv[8];
#pragma unroll
  for (int r = 0; r < 8; r++) inv[r] = 1.0f / lrun[r];
#pragma unroll
  for (int t = 0; t < DT; t++)
#pragma unroll
    for (int r = 0; r < 8; r++) {
      int p = rowbase + r + hi * 8;
      int d = t * 16 + l15;
      int n = (h0 + p / wsp) * RESI + (w0 + p % wsp);
      size_t g = ((size_t)b * NTOK + n) * C + head * DH + d;
      float val = oacc[t][r] * inv[r];
      if (OUT_BF16) ((__bf16*)outp)[g] = (__bf16)val;
      else          ((float*)outp)[g] = val;
    }
}

// ---------------------------------------------------------------------------
// LePE depthwise 3x3 conv on windowed v (hsp=8, wsp=16, zero pad per window),
// accumulated into sy.
// ---------------------------------------------------------------------------
__global__ void lepe_conv_add_kernel(const __bf16* __restrict__ v, const float* __restrict__ w,
                                     const float* __restrict__ bias, float* __restrict__ sy) {
  size_t idx = (size_t)blockIdx.x * 256 + threadIdx.x;
  if (idx >= (size_t)64 * 256 * 256) return;
  int c = (int)(idx & 255);
  size_t t = idx >> 8;
  int n = (int)(t & 255);
  int b = (int)(t >> 8);
  int h = n >> 4, wc = n & 15;
  int wh = h >> 3, i = h & 7;  // hsp = 8
  float acc = bias[c];
#pragma unroll
  for (int di = 0; di < 3; di++) {
    int ii = i + di - 1;
    if (ii < 0 || ii >= 8) continue;
#pragma unroll
    for (int dj = 0; dj < 3; dj++) {
      int jj = wc + dj - 1;
      if (jj < 0 || jj >= 16) continue;  // wsp = 16
      int nn = ((wh * 8 + ii) << 4) + jj;
      acc += w[c * 9 + di * 3 + dj] * (float)v[((size_t)b * 256 + nn) * 256 + c];
    }
  }
  sy[idx] += acc;
}

// ---------------------------------------------------------------------------
extern "C" void kernel_launch(void* const* d_in, const int* in_sizes, int n_in,
                              void* d_out, int out_size, void* d_ws, size_t ws_size,
                              hipStream_t stream) {
  (void)in_sizes; (void)n_in; (void)out_size; (void)ws_size;
  constexpr int C = 256, NTOK = 256, B = 64;
  constexpr int M = B * NTOK;  // 16384 token rows

  const float* hx = (const float*)d_in[0];
  const float* hy = (const float*)d_in[1];
  const float* W[8];  for (int i = 0; i < 8; i++) W[i] = (const float*)d_in[2 + i];
  const float* bb[8]; for (int i = 0; i < 8; i++) bb[i] = (const float*)d_in[10 + i];
  // W/bb order: q, k, v, o, qd, kd, vd, od
  const float* lepe1_w = (const float*)d_in[20];
  const float* lepe1_b = (const float*)d_in[21];
  const float* ln1x_g = (const float*)d_in[22]; const float* ln1x_b = (const float*)d_in[23];
  const float* ln1y_g = (const float*)d_in[24]; const float* ln1y_b = (const float*)d_in[25];
  const float* ln2x_g = (const float*)d_in[26]; const float* ln2x_b = (const float*)d_in[27];
  const float* ln2y_g = (const float*)d_in[28]; const float* ln2y_b = (const float*)d_in[29];
  const float* mx_W1 = (const float*)d_in[30]; const float* mx_b1 = (const float*)d_in[31];
  const float* mx_W2 = (const float*)d_in[32]; const float* mx_b2 = (const float*)d_in[33];
  const float* my_W1 = (const float*)d_in[34]; const float* my_b1 = (const float*)d_in[35];
  const float* my_W2 = (const float*)d_in[36]; const float* my_b2 = (const float*)d_in[37];
  const float* w11 = (const float*)d_in[38]; const float* w12 = (const float*)d_in[39];
  const float* w21 = (const float*)d_in[40]; const float* w22 = (const float*)d_in[41];

  // ---- workspace carve (256B aligned bump allocator) ----
  size_t off = 0;
  auto take = [&](size_t bytes) -> void* {
    void* p = (char*)d_ws + off;
    off += (bytes + 255) & ~(size_t)255;
    return p;
  };
  __bf16* Wt[8]; for (int i = 0; i < 8; i++) Wt[i] = (__bf16*)take((size_t)C * C * 2);
  __bf16* WtX1 = (__bf16*)take((size_t)1024 * C * 2);
  __bf16* WtX2 = (__bf16*)take((size_t)1024 * C * 2);
  __bf16* WtY1 = (__bf16*)take((size_t)1024 * C * 2);
  __bf16* WtY2 = (__bf16*)take((size_t)1024 * C * 2);
  __bf16* xn = (__bf16*)take((size_t)M * C * 2);
  __bf16* yn = (__bf16*)take((size_t)M * C * 2);
  __bf16* qB  = (__bf16*)take((size_t)M * C * 2);
  __bf16* kB  = (__bf16*)take((size_t)M * C * 2);
  __bf16* vB  = (__bf16*)take((size_t)M * C * 2);
  __bf16* qdB = (__bf16*)take((size_t)M * C * 2);
  __bf16* kdB = (__bf16*)take((size_t)M * C * 2);
  __bf16* vdB = (__bf16*)take((size_t)M * C * 2);
  __bf16* ctxx = (__bf16*)take((size_t)M * C * 2);
  __bf16* ctxy = (__bf16*)take((size_t)M * C * 2);
  float* sx = (float*)take((size_t)M * C * 4);
  float* sy = (float*)take((size_t)M * C * 4);
  // MLP hidden buffers alias the (dead by then) qkv/ctx region: each 32MB.
  __bf16* h1x = qB;   // overlays q,k,v,qd (4 x 8MB, contiguous)
  __bf16* h1y = kdB;  // overlays kd,vd,ctxx,ctxy

  // ---- 1. weight prep: cast + transpose to bf16 (N-major, K-contiguous) ----
  for (int i = 0; i < 8; i++)
    transpose_cast_kernel<<<dim3((C * C + 255) / 256), dim3(256), 0, stream>>>(W[i], Wt[i], C, C);
  transpose_cast_kernel<<<dim3((C * 1024 + 255) / 256), dim3(256), 0, stream>>>(mx_W1, WtX1, C, 1024);
  transpose_cast_kernel<<<dim3((C * 1024 + 255) / 256), dim3(256), 0, stream>>>(mx_W2, WtX2, 1024, C);
  transpose_cast_kernel<<<dim3((C * 1024 + 255) / 256), dim3(256), 0, stream>>>(my_W1, WtY1, C, 1024);
  transpose_cast_kernel<<<dim3((C * 1024 + 255) / 256), dim3(256), 0, stream>>>(my_W2, WtY2, 1024, C);

  // ---- 2. LN1 + cast ----
  ln_cast_kernel<<<dim3(M), dim3(256), 0, stream>>>(hx, ln1x_g, ln1x_b, xn);
  ln_cast_kernel<<<dim3(M), dim3(256), 0, stream>>>(hy, ln1y_g, ln1y_b, yn);

  // ---- 3. QKV projections (bf16 outputs) ----
  dim3 gC(C / 64, M / 128);
  gemm128x64<1><<<gC, dim3(256), 0, stream>>>(xn, C, Wt[0], C, bb[0], C, (void*)qB,  nullptr, nullptr, nullptr, nullptr);
  gemm128x64<1><<<gC, dim3(256), 0, stream>>>(xn, C, Wt[1], C, bb[1], C, (void*)kB,  nullptr, nullptr, nullptr, nullptr);
  gemm128x64<1><<<gC, dim3(256), 0, stream>>>(xn, C, Wt[2], C, bb[2], C, (void*)vB,  nullptr, nullptr, nullptr, nullptr);
  gemm128x64<1><<<gC, dim3(256), 0, stream>>>(yn, C, Wt[4], C, bb[4], C, (void*)qdB, nullptr, nullptr, nullptr, nullptr);
  gemm128x64<1><<<gC, dim3(256), 0, stream>>>(yn, C, Wt[5], C, bb[5], C, (void*)kdB, nullptr, nullptr, nullptr, nullptr);
  gemm128x64<1><<<gC, dim3(256), 0, stream>>>(yn, C, Wt[6], C, bb[6], C, (void*)vdB, nullptr, nullptr, nullptr, nullptr);

  // ---- 4. attention (cross-stream) ----
  const float sc_g = 0.17677669529663687f;  // 1/sqrt(32)
  const float sc_l = 0.125f;                // 1/sqrt(64)
  attn_kernel<32, 256, true ><<<dim3(B * 8), dim3(512), 0, stream>>>(qB,  kdB, vdB, (void*)ctxx, 8, 16, 16, sc_g);
  attn_kernel<32, 256, true ><<<dim3(B * 8), dim3(512), 0, stream>>>(qdB, kB,  vB,  (void*)ctxy, 8, 16, 16, sc_g);
  attn_kernel<64, 128, false><<<dim3(B * 2 * 4), dim3(256), 0, stream>>>(qB,  kdB, vdB, (void*)sx, 4, 16, 8, sc_l);
  attn_kernel<64, 128, false><<<dim3(B * 2 * 4), dim3(256), 0, stream>>>(qdB, kB,  vB,  (void*)sy, 4, 8, 16, sc_l);

  // ---- 5. LePE depthwise conv added into sy (idx==1 branch only) ----
  lepe_conv_add_kernel<<<dim3(((size_t)M * C + 255) / 256), dim3(256), 0, stream>>>(vB, lepe1_w, lepe1_b, sy);

  // ---- 6. output projection fused with combine + residual:
  //      xres = w11*sx + w12*(ctx@Wo + bo) + hx   (written in place into sx)
  gemm128x64<4><<<gC, dim3(256), 0, stream>>>(ctxx, C, Wt[3], C, bb[3], C, (void*)sx, sx, hx, w11, w12);
  gemm128x64<4><<<gC, dim3(256), 0, stream>>>(ctxy, C, Wt[7], C, bb[7], C, (void*)sy, sy, hy, w21, w22);

  // ---- 7. LN2 + cast (reuse xn/yn) ----
  ln_cast_kernel<<<dim3(M), dim3(256), 0, stream>>>(sx, ln2x_g, ln2x_b, xn);
  ln_cast_kernel<<<dim3(M), dim3(256), 0, stream>>>(sy, ln2y_g, ln2y_b, yn);

  // ---- 8. MLP fc1 (GELU -> bf16) ----
  dim3 gH(1024 / 64, M / 128);
  gemm128x64<2><<<gH, dim3(256), 0, stream>>>(xn, C, WtX1, C, mx_b1, 1024, (void*)h1x, nullptr, nullptr, nullptr, nullptr);
  gemm128x64<2><<<gH, dim3(256), 0, stream>>>(yn, C, WtY1, C, my_b1, 1024, (void*)h1y, nullptr, nullptr, nullptr, nullptr);

  // ---- 9. MLP fc2 + residual -> final outputs ----
  float* outx = (float*)d_out;
  float* outy = outx + (size_t)M * C;
  gemm128x64<3><<<gC, dim3(256), 0, stream>>>(h1x, 1024, WtX2, 1024, mx_b2, C, (void*)outx, sx, nullptr, nullptr, nullptr);
  gemm128x64<3><<<gC, dim3(256), 0, stream>>>(h1y, 1024, WtY2, 1024, my_b2, C, (void*)outy, sy, nullptr, nullptr, nullptr);
}
